// ODE_6305011990654
// MI455X (gfx1250) — matrix operand order repaired and checked by reading, not compiled, not run on hardware
//
#include <hip/hip_runtime.h>
#include <hip/hip_bf16.h>

typedef __attribute__((ext_vector_type(16))) _Float16 v16h;
typedef __attribute__((ext_vector_type(8)))  _Float16 v8h;
typedef __attribute__((ext_vector_type(8)))  float    v8f;

// ---------------- workspace layout (bytes) ----------------
// packed f16 weights : [0, 22528)        11264 halves
// gamma  f32         : [22528, 38912)    4*16*64 floats
// beta'  f32         : [38912, 55296)    4*16*64 floats  (beta' = gamma*bias + beta)
#define PACKED_HALVES 11264
#define GAMMA_OFF 22528
#define BETA_OFF  38912

struct FilmArgs {
    const float* t;
    const float* scene;
    const float* Gw[4];
    const float* Gb[4];
    const float* Bw[4];
    const float* Bb[4];
    float* gamma;
    float* beta;
};

struct PackArgs {
    const float* W[4];
    const float* bvec[4];
    _Float16* packed;
    const float* gamma;
    float* beta;
};

// gamma/beta: 2 (sel) * 4 (layer) * 16 (b) * 64 (o) = 8192 items, 129-MAC dot each
__global__ __launch_bounds__(256) void film_setup(FilmArgs fa) {
    int u = blockIdx.x * 256 + threadIdx.x;
    if (u >= 8192) return;
    int sel   = u >> 12;         // 0 = gamma, 1 = beta
    int r     = u & 4095;
    int layer = r >> 10;
    int b     = (r >> 6) & 15;
    int o     = r & 63;
    int dout  = (layer == 3) ? 2 : 64;
    float res = 0.0f;
    if (o < dout) {
        const float* W  = sel ? fa.Bw[layer] : fa.Gw[layer];
        const float* bb = sel ? fa.Bb[layer] : fa.Gb[layer];
        float acc = bb[o];
        acc += fa.t[0] * W[o];                      // k = 0 -> t
        for (int k = 1; k < 129; ++k)
            acc += fa.scene[b * 128 + (k - 1)] * W[k * dout + o];
        res = acc;
    }
    float* dst = sel ? fa.beta : fa.gamma;
    dst[layer * 1024 + b * 64 + o] = res;
}

// Pack Ws into CDNA5 WMMA B-matrix (32x16 f16) lane layout:
//   lane L: group g=L/16, column n=L%16; half j<8 holds K = kc*32 + g*8 + j, j>=8 holds K = kc*32 + 16 + g*8 + (j-8)
// tile order per layer: tile = kc*NT + nt ; 512 halves per tile.
// Also folds bias into beta: beta' = gamma*bias + beta (film_setup already done).
__global__ __launch_bounds__(256) void pack_weights(PackArgs pa) {
    int p = blockIdx.x * 256 + threadIdx.x;
    if (p < PACKED_HALVES) {
        int layer, off;
        if      (p < 2048)  { layer = 0; off = 0;     }
        else if (p < 6144)  { layer = 1; off = 2048;  }
        else if (p < 10240) { layer = 2; off = 6144;  }
        else                { layer = 3; off = 10240; }
        int tloc = p - off;
        int tile = tloc >> 9;
        int q    = tloc & 511;
        int L    = q >> 4;
        int j    = q & 15;
        int NTl  = (layer == 3) ? 1 : 4;
        int kc   = tile / NTl;
        int nt   = tile % NTl;
        int g    = L >> 4;
        int n    = nt * 16 + (L & 15);
        int k    = kc * 32 + ((j < 8) ? g * 8 + j : 16 + g * 8 + (j - 8));
        int Kact = (layer == 0) ? 18 : 64;
        int Nact = (layer == 3) ? 2  : 64;
        float v = (k < Kact && n < Nact) ? pa.W[layer][k * Nact + n] : 0.0f;
        pa.packed[p] = (_Float16)v;
    }
    if (p < 4096) {              // beta' = gamma*bias + beta
        int layer = p >> 10;
        int o     = p & 63;
        int dout  = (layer == 3) ? 2 : 64;
        float bi  = (o < dout) ? pa.bvec[layer][o] : 0.0f;
        pa.beta[p] = pa.beta[p] + pa.gamma[p] * bi;
    }
}

__device__ __forceinline__ v8f wmma16(v16h a, v16h bm, v8f c) {
    return __builtin_amdgcn_wmma_f32_16x16x32_f16(false, a, false, bm, (short)0, c,
                                                  false, false);
}

// Branch-free tanh: hardware v_tanh_f32 when available, else exp2+rcp (4 VALU ops).
__device__ __forceinline__ float fast_tanh(float x) {
#if __has_builtin(__builtin_amdgcn_tanhf)
    return __builtin_amdgcn_tanhf(x);
#else
    float xc = fminf(fmaxf(x, -9.0f), 9.0f);
    float e  = __builtin_amdgcn_exp2f(xc * 2.88539008177792681472f); // 2*log2(e)
    return (e - 1.0f) * __builtin_amdgcn_rcpf(e + 1.0f);
#endif
}

__device__ __forceinline__ v16h load_a_chunk(const _Float16* wact, int m, int g, int c) {
    // A 16x32 f16 layout: lane group g, halves j<8 -> K = c*32 + g*8 + j,
    //                                  halves j>=8 -> K = c*32 + 16 + g*8 + (j-8)
    v8h lo = *(const v8h*)(wact + m * 64 + c * 32 + g * 8);
    v8h hi = *(const v8h*)(wact + m * 64 + c * 32 + 16 + g * 8);
    return __builtin_shufflevector(lo, hi, 0, 1, 2, 3, 4, 5, 6, 7,
                                   8, 9, 10, 11, 12, 13, 14, 15);
}

#define TILES_PER_WAVE 8

// Main kernel: 8 waves/block, 8 row-tiles (16 rows each) per wave.
// Full 4-layer FiLM-MLP in WMMA; weights LDS-resident per block (async DMA'd in).
__global__ __launch_bounds__(256) void ode_main(const float* __restrict__ h,
                                                const _Float16* __restrict__ packed,
                                                const float* __restrict__ gammaA,
                                                const float* __restrict__ betaA,
                                                float* __restrict__ out) {
    __shared__ __attribute__((aligned(32))) uint4 smem[2432];   // 38912 B
    _Float16* w_lds = (_Float16*)smem;                          // 11264 halves
    _Float16* actb  = (_Float16*)smem + PACKED_HALVES;          // 8 waves * 1024 halves

    const int tid = threadIdx.x;
    // Async global->LDS DMA of the 22528-B packed weight image (ASYNCcnt path):
    // each lane copies 16 B straight into LDS, no VGPR staging.
    {
        const uint4* pw = (const uint4*)packed;
        for (int i = tid; i < 1408; i += 256) {
            unsigned lds_off = (unsigned)(uintptr_t)(const void*)(smem + i); // flat low32 = LDS addr
            unsigned long long ga = (unsigned long long)(uintptr_t)(pw + i);
            asm volatile("global_load_async_to_lds_b128 %0, %1, off"
                         :: "v"(lds_off), "v"(ga) : "memory");
        }
        asm volatile("s_wait_asynccnt 0x0" ::: "memory");
    }
    __syncthreads();

    const int wave = tid >> 5;
    const int lane = tid & 31;
    const int g    = lane >> 4;
    const int nl   = lane & 15;
    const int waveTileBase = (blockIdx.x * 8 + wave) * TILES_PER_WAVE;
    // a wave covers 128 consecutive rows; 65536 rows per batch element -> b constant
    const int b = (waveTileBase << 4) >> 16;
    _Float16* wact = actb + wave * 1024;

    // ---- hoisted FiLM parameters (constant over this wave's tiles) ----
    float gaf[3][4], bef[3][4];
#pragma unroll
    for (int layer = 0; layer < 3; ++layer)
#pragma unroll
        for (int nt = 0; nt < 4; ++nt) {
            int col = nt * 16 + nl;
            gaf[layer][nt] = gammaA[layer * 1024 + b * 64 + col];
            bef[layer][nt] = betaA [layer * 1024 + b * 64 + col];
        }
    const float ga3 = gammaA[3 * 1024 + b * 64 + nl];
    const float be3 = betaA [3 * 1024 + b * 64 + nl];

    const int LOFF[4] = {0, 2048, 6144, 10240};
    const v8f vzero = {0.f, 0.f, 0.f, 0.f, 0.f, 0.f, 0.f, 0.f};

    for (int it = 0; it < TILES_PER_WAVE; ++it) {
        const int rowBase = (waveTileBase + it) << 4;

        // ---- layer 0 input: Fourier features, built directly in A layout ----
        const float2 hv = ((const float2*)h)[rowBase + nl];
        float s0[4], c0[4], s1[4], c1[4];
#pragma unroll
        for (int i = 0; i < 4; ++i) {
            float f = 3.14159265358979323846f * (float)(1 << i);
            s0[i] = __sinf(f * hv.x); c0[i] = __cosf(f * hv.x);
            s1[i] = __sinf(f * hv.y); c1[i] = __cosf(f * hv.y);
        }
        float feat[18] = {hv.x, hv.y,
                          s0[0], s1[0], c0[0], c1[0],
                          s0[1], s1[1], c0[1], c1[1],
                          s0[2], s1[2], c0[2], c1[2],
                          s0[3], s1[3], c0[3], c1[3]};
        v16h a0;
#pragma unroll
        for (int j = 0; j < 8; ++j)
            a0[j] = (_Float16)(g ? feat[8 + j] : feat[j]);
#pragma unroll
        for (int j = 0; j < 8; ++j) {
            float hiv = 0.0f;
            if (j < 2) hiv = g ? 0.0f : feat[16 + j];   // K=16,17 only valid for g=0
            a0[8 + j] = (_Float16)hiv;
        }

        // ---- layer 0: K=32 (1 chunk), N=64 (4 tiles) ----
        v8f acc[4];
#pragma unroll
        for (int nt = 0; nt < 4; ++nt) {
            v16h bm = *(const v16h*)(w_lds + LOFF[0] + nt * 512 + lane * 16);
            acc[nt] = wmma16(a0, bm, vzero);
        }
#pragma unroll
        for (int nt = 0; nt < 4; ++nt) {
            int col = nt * 16 + nl;
#pragma unroll
            for (int r = 0; r < 8; ++r) {
                float v = fast_tanh(fmaf(gaf[0][nt], acc[nt][r], bef[0][nt]));
                wact[(g * 8 + r) * 64 + col] = (_Float16)v;
            }
        }

        // ---- layers 1,2: K=64 (2 chunks), N=64 (4 tiles), FiLM + tanh ----
#pragma unroll
        for (int layer = 1; layer <= 2; ++layer) {
            v16h aL = load_a_chunk(wact, nl, g, 0);
            v16h aH = load_a_chunk(wact, nl, g, 1);
#pragma unroll
            for (int nt = 0; nt < 4; ++nt) {
                v16h b0 = *(const v16h*)(w_lds + LOFF[layer] + (0 * 4 + nt) * 512 + lane * 16);
                v16h b1 = *(const v16h*)(w_lds + LOFF[layer] + (1 * 4 + nt) * 512 + lane * 16);
                v8f z  = wmma16(aL, b0, vzero);
                acc[nt] = wmma16(aH, b1, z);
            }
#pragma unroll
            for (int nt = 0; nt < 4; ++nt) {
                int col = nt * 16 + nl;
#pragma unroll
                for (int r = 0; r < 8; ++r) {
                    float v = fast_tanh(fmaf(gaf[layer][nt], acc[nt][r], bef[layer][nt]));
                    wact[(g * 8 + r) * 64 + col] = (_Float16)v;
                }
            }
        }

        // ---- layer 3: K=64 (2 chunks), N=16 (2 valid cols), FiLM, no tanh ----
        {
            v16h aL = load_a_chunk(wact, nl, g, 0);
            v16h aH = load_a_chunk(wact, nl, g, 1);
            v16h b0 = *(const v16h*)(w_lds + LOFF[3] + 0 * 512 + lane * 16);
            v16h b1 = *(const v16h*)(w_lds + LOFF[3] + 1 * 512 + lane * 16);
            v8f z = wmma16(aL, b0, vzero);
            v8f o = wmma16(aH, b1, z);
            if (nl < 2) {
#pragma unroll
                for (int r = 0; r < 8; ++r) {
                    float v = fmaf(ga3, o[r], be3);
                    out[(rowBase + g * 8 + r) * 2 + nl] = v;
                }
            }
        }
    }
}

extern "C" void kernel_launch(void* const* d_in, const int* in_sizes, int n_in,
                              void* d_out, int out_size, void* d_ws, size_t ws_size,
                              hipStream_t stream) {
    (void)in_sizes; (void)n_in; (void)out_size; (void)ws_size;

    const float* t     = (const float*)d_in[0];
    const float* h     = (const float*)d_in[1];
    const float* scene = (const float*)d_in[2];

    char* ws = (char*)d_ws;
    _Float16* packed = (_Float16*)ws;
    float* gammaA = (float*)(ws + GAMMA_OFF);
    float* betaA  = (float*)(ws + BETA_OFF);

    FilmArgs fa;
    fa.t = t; fa.scene = scene;
    for (int i = 0; i < 4; ++i) {
        fa.Gw[i] = (const float*)d_in[11 + i];
        fa.Gb[i] = (const float*)d_in[15 + i];
        fa.Bw[i] = (const float*)d_in[19 + i];
        fa.Bb[i] = (const float*)d_in[23 + i];
    }
    fa.gamma = gammaA; fa.beta = betaA;
    film_setup<<<32, 256, 0, stream>>>(fa);

    PackArgs pa;
    for (int i = 0; i < 4; ++i) {
        pa.W[i]    = (const float*)d_in[3 + i];
        pa.bvec[i] = (const float*)d_in[7 + i];
    }
    pa.packed = packed; pa.gamma = gammaA; pa.beta = betaA;
    pack_weights<<<44, 256, 0, stream>>>(pa);

    // 65536 row-tiles of 16 rows; 8 tiles per wave, 8 waves per block -> 1024 blocks
    ode_main<<<1024, 256, 0, stream>>>(h, packed, gammaA, betaA, (float*)d_out);
}
